// Attention_5849745457833
// MI455X (gfx1250) — matrix operand order repaired and checked by reading, not compiled, not run on hardware
//
#include <hip/hip_runtime.h>
#include <hip/hip_bf16.h>

// ---------------------------------------------------------------------------
// Fused Bahdanau-style attention for MI455X (gfx1250, wave32, WMMA bf16)
//   psi = h_enc @ W_psi^T + b_psi         [B,T,A]   (bf16 ws)
//   phi = h_dec @ W_phi^T + b_phi         [B,Q,A]   (bf16 ws, permuted)
//   e   = psi @ phi^T  per batch          [B,T,Q]   (f32 ws)
//   a   = softmax_T(e)                    [B,Q,T]   (bf16 ws, transposed)
//   c   = a @ h_enc    per batch          [B,Q,H]   (f32 out)
// All matmuls use v_wmma_f32_16x16x32_bf16 (f32 accumulate). k_scores uses
// double-buffered GLOBAL_LOAD_ASYNC_TO_LDS_B128 (ASYNCcnt) for copy/compute
// overlap since its operands are already bf16 in the workspace.
// ---------------------------------------------------------------------------

typedef __attribute__((ext_vector_type(16))) __bf16 v16bf;
typedef __attribute__((ext_vector_type(8)))  __bf16 v8bf;
typedef __attribute__((ext_vector_type(8)))  float  v8f;

#define BATCH 16
#define TLEN  2048
#define QLEN  64
#define HDIM  1024
#define ADIM  1024

static __device__ __forceinline__ v8f wmma_bf16(v16bf a, v16bf b, v8f c) {
  return __builtin_amdgcn_wmma_f32_16x16x32_bf16(
      /*neg_a=*/false, a, /*neg_b=*/false, b,
      /*c_mod=*/(short)0, c, /*reuse_a=*/false, /*reuse_b=*/false);
}

// A fragment (16x32 bf16, MxK): lane half h holds K = [8h,8h+8) and [16+8h,16+8h+8)
static __device__ __forceinline__ v16bf load_frag_a(const __bf16* rowp, int half) {
  union { v16bf v; v8bf h[2]; } f;
  f.h[0] = *(const v8bf*)(rowp + half * 8);
  f.h[1] = *(const v8bf*)(rowp + 16 + half * 8);
  return f.v;
}

// B fragment (32x16 bf16, KxN, stored N-major [n][k]): lane half h holds K = [8h,8h+8) and [16+8h,16+8h+8)
static __device__ __forceinline__ v16bf load_frag_b(const __bf16* rowp, int half) {
  union { v16bf v; v8bf h[2]; } f;
  f.h[0] = *(const v8bf*)(rowp + half * 8);
  f.h[1] = *(const v8bf*)(rowp + 16 + half * 8);
  return f.v;
}

// Async 16B global -> LDS copy (gfx1250, ASYNCcnt-tracked, no VGPR data path).
// VDST VGPR carries the LDS byte offset (low 32 bits of the flat LDS address).
static __device__ __forceinline__ void async_ld_b128(unsigned ldsOff, const void* gptr) {
  asm volatile("global_load_async_to_lds_b128 %0, %1, off"
               :: "v"(ldsOff), "v"((unsigned long long)(size_t)gptr)
               : "memory");
}
static __device__ __forceinline__ void wait_async0() {
  asm volatile("s_wait_asynccnt 0x0" ::: "memory");
}

// ---------------------------------------------------------------------------
// Kernel 1/2: projection GEMM  out[m,n] = sum_k X[m,k]*W[n,k] + bias[n]
// BM=128 BN=128 BK=32, 8 waves (4 M x 2 N), wave tile 32x64 (2x4 frags).
// OUT_PERM==1 reshuffles rows m=q*B+b of h_dec into [B,Q,A] layout.
// ---------------------------------------------------------------------------
template <int OUT_PERM>
__global__ __launch_bounds__(256) void k_gemm_proj(
    const float* __restrict__ X, const float* __restrict__ W,
    const float* __restrict__ bias, __bf16* __restrict__ out,
    int M, int N, int K) {
  __shared__ __align__(16) __bf16 lA[128 * 32];
  __shared__ __align__(16) __bf16 lB[128 * 32];

  const int tid   = threadIdx.x;
  const int lane  = tid & 31;
  const int wave  = tid >> 5;
  const int waveM = wave >> 1;       // 0..3
  const int waveN = wave & 1;        // 0..1
  const int col16 = lane & 15;
  const int half  = lane >> 4;

  const int mBase = blockIdx.y * 128;
  const int nBase = blockIdx.x * 128;

  v8f acc[2][4] = {};

  const int ldrow = tid >> 1;          // 0..127
  const int ldcb  = (tid & 1) * 16;    // 0 / 16
  const float* pa = X + (size_t)(mBase + ldrow) * K + ldcb;
  const float* pb = W + (size_t)(nBase + ldrow) * K + ldcb;

  for (int k0 = 0; k0 < K; k0 += 32) {
    __align__(16) __bf16 ta[16], tb[16];
#pragma unroll
    for (int i = 0; i < 4; ++i) {
      float4 fa = *(const float4*)(pa + k0 + 4 * i);
      float4 fb = *(const float4*)(pb + k0 + 4 * i);
      ta[4 * i + 0] = (__bf16)fa.x; ta[4 * i + 1] = (__bf16)fa.y;
      ta[4 * i + 2] = (__bf16)fa.z; ta[4 * i + 3] = (__bf16)fa.w;
      tb[4 * i + 0] = (__bf16)fb.x; tb[4 * i + 1] = (__bf16)fb.y;
      tb[4 * i + 2] = (__bf16)fb.z; tb[4 * i + 3] = (__bf16)fb.w;
    }
    // prefetch next K-tile (global_prefetch_b8; speculative, OOB safe)
    __builtin_prefetch(pa + k0 + 32, 0, 0);
    __builtin_prefetch(pb + k0 + 32, 0, 0);

    __syncthreads();
    *(v8bf*)&lA[ldrow * 32 + ldcb]     = *(const v8bf*)&ta[0];
    *(v8bf*)&lA[ldrow * 32 + ldcb + 8] = *(const v8bf*)&ta[8];
    *(v8bf*)&lB[ldrow * 32 + ldcb]     = *(const v8bf*)&tb[0];
    *(v8bf*)&lB[ldrow * 32 + ldcb + 8] = *(const v8bf*)&tb[8];
    __syncthreads();

    v16bf af[2], bfm[4];
#pragma unroll
    for (int fm = 0; fm < 2; ++fm)
      af[fm] = load_frag_a(&lA[(waveM * 32 + fm * 16 + col16) * 32], half);
#pragma unroll
    for (int fn = 0; fn < 4; ++fn)
      bfm[fn] = load_frag_b(&lB[(waveN * 64 + fn * 16 + col16) * 32], half);
#pragma unroll
    for (int fm = 0; fm < 2; ++fm)
#pragma unroll
      for (int fn = 0; fn < 4; ++fn)
        acc[fm][fn] = wmma_bf16(af[fm], bfm[fn], acc[fm][fn]);
  }

  // epilogue: bias + bf16 store
#pragma unroll
  for (int fn = 0; fn < 4; ++fn) {
    const int n = nBase + waveN * 64 + fn * 16 + col16;
    const float bv = bias[n];
#pragma unroll
    for (int fm = 0; fm < 2; ++fm) {
#pragma unroll
      for (int r = 0; r < 8; ++r) {
        const int m = mBase + waveM * 32 + fm * 16 + half * 8 + r;
        const float v = acc[fm][fn][r] + bv;
        size_t oi;
        if (OUT_PERM) {  // row m = q*B + b  ->  [b][q][n]
          oi = ((size_t)((m & 15) * QLEN + (m >> 4))) * (size_t)N + n;
        } else {
          oi = (size_t)m * (size_t)N + n;
        }
        out[oi] = (__bf16)v;
      }
    }
  }
}

// ---------------------------------------------------------------------------
// Kernel 3: scores  e[b,t,q] = sum_a psi[b,t,a] * phi[b,q,a]
// per batch: M=T=2048, N=Q=64, K=A=1024.  BM=128 BN=64 BK=32.
// 8 waves (4 M x 2 N), wave tile 32x32 (2x2 frags).
// Double-buffered LDS filled by GLOBAL_LOAD_ASYNC_TO_LDS_B128: the async
// engine fills buffer k^1 while WMMA consumes buffer k.
// ---------------------------------------------------------------------------
__global__ __launch_bounds__(256) void k_scores(
    const __bf16* __restrict__ psi, const __bf16* __restrict__ phi,
    float* __restrict__ e) {
  __shared__ __align__(16) __bf16 lA[2][128 * 32];  // 2 x 8 KB
  __shared__ __align__(16) __bf16 lB[2][64 * 32];   // 2 x 4 KB

  const int tid   = threadIdx.x;
  const int lane  = tid & 31;
  const int wave  = tid >> 5;
  const int waveM = wave >> 1;
  const int waveN = wave & 1;
  const int col16 = lane & 15;
  const int half  = lane >> 4;

  const int b     = blockIdx.y;
  const int tBase = blockIdx.x * 128;
  const __bf16* pA = psi + (size_t)b * TLEN * ADIM;
  const __bf16* pB = phi + (size_t)b * QLEN * ADIM;

  v8f acc[2][2] = {};

  const int arow = tid >> 2;        // 0..63 (A tile uses arow and arow+64)
  const int ac4e = (tid & 3) * 8;   // bf16 element offset within a 32-elem row

  // per-thread global source pointers (advance by k0 elements each step)
  const __bf16* gA0 = pA + (size_t)(tBase + arow) * ADIM + ac4e;
  const __bf16* gA1 = pA + (size_t)(tBase + arow + 64) * ADIM + ac4e;
  const __bf16* gB0 = pB + (size_t)arow * ADIM + ac4e;

  // per-thread LDS byte offsets (buffer 0); buffer 1 = +stride
  const unsigned offA0 = (unsigned)(size_t)&lA[0][arow * 32 + ac4e];
  const unsigned offA1 = (unsigned)(size_t)&lA[0][(arow + 64) * 32 + ac4e];
  const unsigned offB0 = (unsigned)(size_t)&lB[0][arow * 32 + ac4e];
  const unsigned strA  = (unsigned)(128 * 32 * sizeof(__bf16));  // 8192
  const unsigned strB  = (unsigned)(64 * 32 * sizeof(__bf16));   // 4096

  // prologue: async-fill buffer 0 with k0 = 0
  async_ld_b128(offA0, gA0);
  async_ld_b128(offA1, gA1);
  async_ld_b128(offB0, gB0);

  for (int k0 = 0; k0 < ADIM; k0 += 32) {
    const int cur = (k0 >> 5) & 1;
    wait_async0();        // this wave's fills of buffer `cur` have landed
    __syncthreads();      // all waves' fills have landed; prev reads done

    if (k0 + 32 < ADIM) { // async-fill the other buffer while we compute
      const unsigned bo = (unsigned)(cur ^ 1);
      async_ld_b128(offA0 + bo * strA, gA0 + k0 + 32);
      async_ld_b128(offA1 + bo * strA, gA1 + k0 + 32);
      async_ld_b128(offB0 + bo * strB, gB0 + k0 + 32);
    }

    v16bf af[2], bfm[2];
#pragma unroll
    for (int fm = 0; fm < 2; ++fm)
      af[fm] = load_frag_a(&lA[cur][(waveM * 32 + fm * 16 + col16) * 32], half);
#pragma unroll
    for (int fn = 0; fn < 2; ++fn)
      bfm[fn] = load_frag_b(&lB[cur][(waveN * 32 + fn * 16 + col16) * 32], half);
#pragma unroll
    for (int fm = 0; fm < 2; ++fm)
#pragma unroll
      for (int fn = 0; fn < 2; ++fn)
        acc[fm][fn] = wmma_bf16(af[fm], bfm[fn], acc[fm][fn]);
  }

#pragma unroll
  for (int fm = 0; fm < 2; ++fm)
#pragma unroll
    for (int fn = 0; fn < 2; ++fn)
#pragma unroll
      for (int r = 0; r < 8; ++r) {
        const int t = tBase + waveM * 32 + fm * 16 + half * 8 + r;
        const int q = waveN * 32 + fn * 16 + col16;
        e[((size_t)b * TLEN + t) * QLEN + q] = acc[fm][fn][r];
      }
}

// ---------------------------------------------------------------------------
// Kernel 4: softmax over T for each (b,q); writes attn transposed [B,Q,T] bf16
// ---------------------------------------------------------------------------
__global__ __launch_bounds__(256) void k_softmax(
    const float* __restrict__ e, __bf16* __restrict__ attn) {
  __shared__ float red[256];
  const int q = blockIdx.x;
  const int b = blockIdx.y;
  const int tid = threadIdx.x;

  const float* col = e + (size_t)b * TLEN * QLEN + q;
  float v[8];
  float m = -3.4e38f;
#pragma unroll
  for (int i = 0; i < 8; ++i) {
    v[i] = col[(size_t)(tid + i * 256) * QLEN];
    m = fmaxf(m, v[i]);
  }
  red[tid] = m;
  __syncthreads();
  for (int s = 128; s > 0; s >>= 1) {
    if (tid < s) red[tid] = fmaxf(red[tid], red[tid + s]);
    __syncthreads();
  }
  m = red[0];
  __syncthreads();

  float sum = 0.f;
#pragma unroll
  for (int i = 0; i < 8; ++i) {
    v[i] = __expf(v[i] - m);
    sum += v[i];
  }
  red[tid] = sum;
  __syncthreads();
  for (int s = 128; s > 0; s >>= 1) {
    if (tid < s) red[tid] += red[tid + s];
    __syncthreads();
  }
  const float inv = 1.0f / red[0];

  __bf16* orow = attn + ((size_t)b * QLEN + q) * TLEN;
#pragma unroll
  for (int i = 0; i < 8; ++i)
    orow[tid + i * 256] = (__bf16)(v[i] * inv);
}

// ---------------------------------------------------------------------------
// Kernel 5: context  c[b,q,h] = sum_t attn[b,q,t] * h_enc[b,t,h]
// per batch: M=Q=64, N=H=1024, K=T=2048.  BM=64 BN=128 BK=32.
// 8 waves (2 M x 4 N), wave tile 32x32 (2x2 frags).
// h_enc f32 tile is transposed + converted into LDS [h][t] bf16.
// ---------------------------------------------------------------------------
__global__ __launch_bounds__(256) void k_context(
    const __bf16* __restrict__ attn, const float* __restrict__ henc,
    float* __restrict__ out) {
  __shared__ __align__(16) __bf16 lA[64 * 32];    // [q][t]
  __shared__ __align__(16) __bf16 lB[128 * 32];   // [h][t]

  const int tid   = threadIdx.x;
  const int lane  = tid & 31;
  const int wave  = tid >> 5;
  const int waveM = wave & 1;       // 0..1
  const int waveN = wave >> 1;      // 0..3
  const int col16 = lane & 15;
  const int half  = lane >> 4;

  const int b     = blockIdx.y;
  const int hBase = blockIdx.x * 128;
  const __bf16* pA = attn + (size_t)b * QLEN * TLEN;
  const float*  pH = henc + (size_t)b * TLEN * HDIM;

  v8f acc[2][2] = {};

  const int arow = tid >> 2;          // 0..63
  const int ac4  = (tid & 3) * 8;
  const int btt  = tid >> 3;          // 0..31  (t-row within tile)
  const int bhh  = (tid & 7) * 16;    // h-chunk of 16 floats

  for (int k0 = 0; k0 < TLEN; k0 += 32) {
    uint4 a0 = *(const uint4*)(pA + (size_t)arow * TLEN + k0 + ac4);
    float4 f0 = *(const float4*)(pH + (size_t)(k0 + btt) * HDIM + hBase + bhh + 0);
    float4 f1 = *(const float4*)(pH + (size_t)(k0 + btt) * HDIM + hBase + bhh + 4);
    float4 f2 = *(const float4*)(pH + (size_t)(k0 + btt) * HDIM + hBase + bhh + 8);
    float4 f3 = *(const float4*)(pH + (size_t)(k0 + btt) * HDIM + hBase + bhh + 12);

    __syncthreads();
    *(uint4*)&lA[arow * 32 + ac4] = a0;
    lB[(bhh + 0)  * 32 + btt] = (__bf16)f0.x;
    lB[(bhh + 1)  * 32 + btt] = (__bf16)f0.y;
    lB[(bhh + 2)  * 32 + btt] = (__bf16)f0.z;
    lB[(bhh + 3)  * 32 + btt] = (__bf16)f0.w;
    lB[(bhh + 4)  * 32 + btt] = (__bf16)f1.x;
    lB[(bhh + 5)  * 32 + btt] = (__bf16)f1.y;
    lB[(bhh + 6)  * 32 + btt] = (__bf16)f1.z;
    lB[(bhh + 7)  * 32 + btt] = (__bf16)f1.w;
    lB[(bhh + 8)  * 32 + btt] = (__bf16)f2.x;
    lB[(bhh + 9)  * 32 + btt] = (__bf16)f2.y;
    lB[(bhh + 10) * 32 + btt] = (__bf16)f2.z;
    lB[(bhh + 11) * 32 + btt] = (__bf16)f2.w;
    lB[(bhh + 12) * 32 + btt] = (__bf16)f3.x;
    lB[(bhh + 13) * 32 + btt] = (__bf16)f3.y;
    lB[(bhh + 14) * 32 + btt] = (__bf16)f3.z;
    lB[(bhh + 15) * 32 + btt] = (__bf16)f3.w;
    __syncthreads();

    v16bf af[2], bfm[2];
#pragma unroll
    for (int fm = 0; fm < 2; ++fm)
      af[fm] = load_frag_a(&lA[(waveM * 32 + fm * 16 + col16) * 32], half);
#pragma unroll
    for (int fn = 0; fn < 2; ++fn)
      bfm[fn] = load_frag_b(&lB[(waveN * 32 + fn * 16 + col16) * 32], half);
#pragma unroll
    for (int fm = 0; fm < 2; ++fm)
#pragma unroll
      for (int fn = 0; fn < 2; ++fn)
        acc[fm][fn] = wmma_bf16(af[fm], bfm[fn], acc[fm][fn]);
  }

#pragma unroll
  for (int fm = 0; fm < 2; ++fm)
#pragma unroll
    for (int fn = 0; fn < 2; ++fn)
#pragma unroll
      for (int r = 0; r < 8; ++r) {
        const int q = waveM * 32 + fm * 16 + half * 8 + r;
        const int h = hBase + waveN * 32 + fn * 16 + col16;
        out[((size_t)b * QLEN + q) * HDIM + h] = acc[fm][fn][r];
      }
}

// ---------------------------------------------------------------------------
extern "C" void kernel_launch(void* const* d_in, const int* in_sizes, int n_in,
                              void* d_out, int out_size, void* d_ws, size_t ws_size,
                              hipStream_t stream) {
  const float* h_enc = (const float*)d_in[0];  // [B,T,H]
  const float* h_dec = (const float*)d_in[1];  // [Q,B,H]
  const float* W_psi = (const float*)d_in[2];  // [A,H]
  const float* b_psi = (const float*)d_in[3];  // [A]
  const float* W_phi = (const float*)d_in[4];  // [A,H]
  const float* b_phi = (const float*)d_in[5];  // [A]
  float* c_out = (float*)d_out;                // [B,Q,H]

  // workspace layout
  char* ws = (char*)d_ws;
  const size_t psi_bytes  = (size_t)BATCH * TLEN * ADIM * 2;  // 64 MB bf16
  const size_t phi_bytes  = (size_t)BATCH * QLEN * ADIM * 2;  //  2 MB bf16
  const size_t e_bytes    = (size_t)BATCH * TLEN * QLEN * 4;  //  8 MB f32
  __bf16* psi_ws  = (__bf16*)ws;
  __bf16* phi_ws  = (__bf16*)(ws + psi_bytes);
  float*  e_ws    = (float*)(ws + psi_bytes + phi_bytes);
  __bf16* attn_ws = (__bf16*)(ws + psi_bytes + phi_bytes + e_bytes);

  // 1) psi = h_enc @ W_psi^T + b_psi   (M=B*T=32768, N=A, K=H)
  k_gemm_proj<0><<<dim3(ADIM / 128, (BATCH * TLEN) / 128), 256, 0, stream>>>(
      h_enc, W_psi, b_psi, psi_ws, BATCH * TLEN, ADIM, HDIM);

  // 2) phi = h_dec @ W_phi^T + b_phi   (M=Q*B=1024), permuted out -> [B,Q,A]
  k_gemm_proj<1><<<dim3(ADIM / 128, (QLEN * BATCH) / 128), 256, 0, stream>>>(
      h_dec, W_phi, b_phi, phi_ws, QLEN * BATCH, ADIM, HDIM);

  // 3) e[b] = psi[b] @ phi[b]^T   (async-LDS double-buffered)
  k_scores<<<dim3(TLEN / 128, BATCH), 256, 0, stream>>>(psi_ws, phi_ws, e_ws);

  // 4) softmax over T -> attn [B,Q,T]
  k_softmax<<<dim3(QLEN, BATCH), 256, 0, stream>>>(e_ws, attn_ws);

  // 5) c[b] = attn[b] @ h_enc[b]
  k_context<<<dim3(HDIM / 128, BATCH), 256, 0, stream>>>(attn_ws, h_enc, c_out);
}
